// DARTSCell_4904852652501
// MI455X (gfx1250) — matrix operand order repaired and checked by reading, not compiled, not run on hardware
//
#include <hip/hip_runtime.h>

// ---------------------------------------------------------------------------
// DARTS RNN cell for MI455X (gfx1250, wave32, WMMA bf16 + TDM async staging)
//   inputs: [1024,64,256] f32, hidden: [1,64,256] f32,
//   W0: [512,512] f32, Ws: [8,256,512] f32
//   out: hiddens [1024,64,256] f32  ++  hiddens[-1] [1,64,256] f32
// ---------------------------------------------------------------------------

typedef __attribute__((ext_vector_type(16))) __bf16 v16bf;
typedef __attribute__((ext_vector_type(8)))  __bf16 v8bf;
typedef __attribute__((ext_vector_type(8)))  float  v8f;
typedef __attribute__((ext_vector_type(4)))  unsigned int u32x4;
typedef __attribute__((ext_vector_type(4)))  int i32x4;
typedef __attribute__((ext_vector_type(8)))  int i32x8;

#define T_STEPS 1024
#define SROW    264                 // LDS row stride (ushort); 528B: 16B-aligned, 4-dw bank skew
#define REG_SZ  (64 * SROW)         // one 64x256 state buffer (padded)
#define PACKED_ELEMS ((size_t)10 * 256 * 512)   // 10 matrices of 256x512 bf16, tile-packed

#if defined(__has_builtin)
#  if __has_builtin(__builtin_amdgcn_tensor_load_to_lds)
#    define HAVE_TDM 1
#  endif
#  if __has_builtin(__builtin_amdgcn_s_wait_tensorcnt)
#    define HAVE_TCNT 1
#  endif
#endif

__device__ __forceinline__ unsigned short f2bf(float f) {
  unsigned int u = __float_as_uint(f);
  u += 0x7FFFu + ((u >> 16) & 1u);            // round-to-nearest-even
  return (unsigned short)(u >> 16);
}
__device__ __forceinline__ float bf2f(unsigned short h) {
  return __uint_as_float(((unsigned int)h) << 16);
}
__device__ __forceinline__ v16bf cat16(v8bf lo, v8bf hi) {
  return __builtin_shufflevector(lo, hi, 0,1,2,3,4,5,6,7,8,9,10,11,12,13,14,15);
}
__device__ __forceinline__ v8f v8f_zero() {
  v8f z = {0.f,0.f,0.f,0.f,0.f,0.f,0.f,0.f};
  return z;
}

// A fragment (16x32 bf16) from row-major LDS: two contiguous 16B runs per lane.
__device__ __forceinline__ v16bf load_a_frag(const unsigned short* base, int mt, int kt,
                                             int laneLo, int laneHi) {
  const unsigned short* p = base + (mt * 16 + laneLo) * SROW + kt * 32 + 8 * laneHi;
  v8bf lo = *reinterpret_cast<const v8bf*>(p);        // ds_load_b128
  v8bf hi = *reinterpret_cast<const v8bf*>(p + 16);   // ds_load_b128
  return cat16(lo, hi);
}

// B fragment (32x16 bf16) from pre-packed global weights: 32B contiguous per lane.
__device__ __forceinline__ v16bf load_b_frag(const unsigned short* __restrict__ packed,
                                             int m, int kt, int ntile, int lane) {
  const unsigned short* p = packed + ((size_t)((m * 256 + kt * 32 + ntile) * 32 + lane)) * 16;
  v8bf lo = *reinterpret_cast<const v8bf*>(p);        // global_load_b128
  v8bf hi = *reinterpret_cast<const v8bf*>(p + 8);    // global_load_b128
  return cat16(lo, hi);
}

// ---------------------------------------------------------------------------
// TDM: one 2-D tile descriptor, 64 rows x 512 cols of 2-byte elements,
// global (row-major, stride 512) -> LDS (contiguous). Issued by one wave.
// ---------------------------------------------------------------------------
__device__ __forceinline__ void tdm_load_xp_tile(unsigned lds_byte_addr,
                                                 const unsigned short* gsrc) {
#if defined(HAVE_TDM)
  unsigned long long ga = (unsigned long long)(size_t)gsrc;
  u32x4 g0;
  g0[0] = 1u;                                        // count=1, user descriptor
  g0[1] = lds_byte_addr;                             // D#.lds_addr
  g0[2] = (unsigned)ga;                              // global_addr[31:0]
  g0[3] = (unsigned)((ga >> 32) & 0x01ffffffu) | (2u << 30);  // addr[56:32], type=2
  i32x8 g1;
  g1[0] = 0x00010000;            // workgroup_mask=0, data_size=1 (2 bytes)
  g1[1] = (int)(512u << 16);     // tensor_dim0[15:0]=512 (bits 63:48)
  g1[2] = (int)(64u  << 16);     // tensor_dim0 hi=0, tensor_dim1[15:0]=64
  g1[3] = (int)(512u << 16);     // tensor_dim1 hi=0, tile_dim0=512
  g1[4] = 64;                    // tile_dim1=64, tile_dim2=0
  g1[5] = 512;                   // tensor_dim0_stride[31:0]=512
  g1[6] = 0;                     // stride hi, tensor_dim1_stride lo
  g1[7] = 0;
  i32x4 z4 = {0, 0, 0, 0};
#if __clang_major__ >= 23
  i32x8 z8 = {0, 0, 0, 0, 0, 0, 0, 0};
  __builtin_amdgcn_tensor_load_to_lds(g0, g1, z4, z4, z8, 0);
#else
  __builtin_amdgcn_tensor_load_to_lds(g0, g1, z4, z4, 0);
#endif
#else
  (void)lds_byte_addr; (void)gsrc;
#endif
}
__device__ __forceinline__ void tdm_wait0() {
#if defined(HAVE_TCNT)
  __builtin_amdgcn_s_wait_tensorcnt(0);
#elif defined(HAVE_TDM)
  asm volatile("s_wait_tensorcnt 0x0" ::: "memory");
#endif
}

// ---------------------------------------------------------------------------
// Kernel 1: pack weights (f32 -> bf16) into per-lane WMMA-B fragment order.
// m=0: W0 rows 0..255 (x part), m=1: W0 rows 256..511 (h part), m=2+i: Ws[i].
// ---------------------------------------------------------------------------
__global__ void pack_weights(const float* __restrict__ W0, const float* __restrict__ Ws,
                             unsigned short* __restrict__ packed) {
  int idx  = blockIdx.x * blockDim.x + threadIdx.x;
  int m    = idx >> 17;
  int rem  = idx & 131071;
  int tile = rem >> 9;
  int li   = rem & 511;
  int L = li >> 4, e = li & 15;
  int kt = tile >> 5, nt = tile & 31;
  int row = kt * 32 + ((L >> 4) << 3) + (e & 7) + ((e >> 3) << 4);
  int col = nt * 16 + (L & 15);
  float v;
  if (m == 0)      v = W0[row * 512 + col];
  else if (m == 1) v = W0[(256 + row) * 512 + col];
  else             v = Ws[((size_t)(m - 2) * 256 + row) * 512 + col];
  packed[idx] = f2bf(v);
}

// ---------------------------------------------------------------------------
// Kernel 2: xproj[t] = x_t @ W0x for all t in parallel (hoisted off the chain).
// ---------------------------------------------------------------------------
__global__ __launch_bounds__(256)
void xproj_kernel(const float* __restrict__ x, const unsigned short* __restrict__ packed,
                  unsigned short* __restrict__ xp) {
  __shared__ unsigned short sX[64 * SROW];
  int t = blockIdx.x;
  int tid = threadIdx.x;
  for (int i = tid; i < 64 * 256; i += 256) {
    int row = i >> 8, col = i & 255;
    sX[row * SROW + col] = f2bf(x[(size_t)t * 16384 + i]);
  }
  __syncthreads();
  int wave = tid >> 5, lane = tid & 31;
  int laneLo = lane & 15, laneHi = lane >> 4;
  for (int mt = 0; mt < 4; ++mt) {
    v8f acc[4];
#pragma unroll
    for (int j = 0; j < 4; ++j) acc[j] = v8f_zero();
#pragma unroll
    for (int kt = 0; kt < 8; ++kt) {
      v16bf a = load_a_frag(sX, mt, kt, laneLo, laneHi);
#pragma unroll
      for (int j = 0; j < 4; ++j) {
        v16bf b = load_b_frag(packed, 0, kt, wave + 8 * j, lane);
        acc[j] = __builtin_amdgcn_wmma_f32_16x16x32_bf16(false, a, false, b,
                                                         (short)0, acc[j], false, false);
      }
    }
#pragma unroll
    for (int j = 0; j < 4; ++j) {
      int nt = wave + 8 * j;
#pragma unroll
      for (int r = 0; r < 8; ++r) {
        int row = mt * 16 + r + 8 * laneHi;
        int col = nt * 16 + laneLo;
        xp[((size_t)t * 64 + row) * 512 + col] = f2bf(acc[j][r]);
      }
    }
  }
}

// ---------------------------------------------------------------------------
// Kernel 3: persistent single-workgroup recurrence (16 waves). Wave w owns the
// column-pair (c cols w*16.., h cols 256+w*16..) for all 4 M-tiles. States
// ping-pong in 2 LDS bf16 buffers; xproj[t+1] is TDM-staged into LDS while
// the 8 chain GEMMs of step t run; concat-mean lives in VGPRs.
// ---------------------------------------------------------------------------
__global__ __launch_bounds__(512)
void recurrence(const float* __restrict__ hidden, const unsigned short* __restrict__ packed,
                const unsigned short* __restrict__ xp, float* __restrict__ out) {
  __shared__ unsigned short sA[2 * REG_SZ];     // two 64x256 bf16 state buffers (67.6 KB)
  __shared__ unsigned short sXP[64 * 512];      // TDM stage for xproj[t] (64 KB)

  static constexpr int PRED[9] = {1, 1, 0, 1, 1, 1, 0, 1, 0};
  static constexpr int WREG[9] = {1, 0, 1, 1, 1, 1, -1, -1, -1};
  static constexpr int ACT[9]  = {0, 1, 1, 0, 1, 1, 2, 1, 1};   // 0=tanh 1=relu 2=id

  int tid = threadIdx.x;
  int wave = tid >> 5, lane = tid & 31;
  int laneLo = lane & 15, laneHi = lane >> 4;
  int p = wave;
  unsigned sxp_lds = (unsigned)(size_t)(void*)&sXP[0];

  // h_prev -> region 1
  for (int i = tid; i < 64 * 256; i += 512) {
    int row = i >> 8, col = i & 255;
    sA[REG_SZ + row * SROW + col] = f2bf(hidden[i]);
  }
#if defined(HAVE_TDM)
  if (wave == 0) { tdm_load_xp_tile(sxp_lds, xp); tdm_wait0(); }
#else
  for (int i = tid; i < 64 * 512; i += 512) sXP[i] = xp[i];
#endif
  __syncthreads();

  v8f sum[4];                                   // concat-mean accumulator (regs)
#pragma unroll
  for (int mt = 0; mt < 4; ++mt) sum[mt] = v8f_zero();

  for (int t = 0; t < T_STEPS; ++t) {
#pragma unroll
    for (int k = 0; k < 9; ++k) {
      const unsigned short* abase = sA + PRED[k] * REG_SZ;
      int wm = (k == 0) ? 1 : (1 + k);          // packed matrix (W0h or Ws[k-1])

      v8f accC[4], accH[4];
      if (k == 0) {
        // init from TDM-staged x@W0x
#pragma unroll
        for (int mt = 0; mt < 4; ++mt) {
#pragma unroll
          for (int r = 0; r < 8; ++r) {
            int row = mt * 16 + r + 8 * laneHi;
            int colc = p * 16 + laneLo;
            accC[mt][r] = bf2f(sXP[row * 512 + colc]);
            accH[mt][r] = bf2f(sXP[row * 512 + 256 + colc]);
          }
        }
      } else {
#pragma unroll
        for (int mt = 0; mt < 4; ++mt) { accC[mt] = v8f_zero(); accH[mt] = v8f_zero(); }
      }

      // GEMM: [64,256] (LDS bf16) x [256,512] (packed bf16)
#pragma unroll
      for (int kt = 0; kt < 8; ++kt) {
        v16bf bc = load_b_frag(packed, wm, kt, p, lane);
        v16bf bh = load_b_frag(packed, wm, kt, p + 16, lane);
#pragma unroll
        for (int mt = 0; mt < 4; ++mt) {
          v16bf a = load_a_frag(abase, mt, kt, laneLo, laneHi);
          accC[mt] = __builtin_amdgcn_wmma_f32_16x16x32_bf16(false, a, false, bc,
                                                             (short)0, accC[mt], false, false);
          accH[mt] = __builtin_amdgcn_wmma_f32_16x16x32_bf16(false, a, false, bh,
                                                             (short)0, accH[mt], false, false);
        }
      }

      // elementwise: s' = s + sigmoid(c) * (act(h) - s); fold concat-mean here
      float snew[4][8];
#pragma unroll
      for (int mt = 0; mt < 4; ++mt) {
#pragma unroll
        for (int r = 0; r < 8; ++r) {
          int row = mt * 16 + r + 8 * laneHi;
          int col = p * 16 + laneLo;
          float c = accC[mt][r], hh = accH[mt][r];
          float sprev = bf2f(abase[row * SROW + col]);
          float sig = 1.0f / (1.0f + __expf(-c));
          float a = (ACT[k] == 0) ? tanhf(hh) : ((ACT[k] == 1) ? fmaxf(hh, 0.0f) : hh);
          float sn = sprev + sig * (a - sprev);
          if (k == 1)      sum[mt][r] = sn;
          else if (k >= 2) sum[mt][r] += sn;
          snew[mt][r] = (k == 8) ? (sum[mt][r] * 0.125f) : sn;   // k=8 -> h_new
        }
      }
      __syncthreads();                          // all pred/stage reads done

      // write phase
      if (k == 0) {
        if (wave == 0 && t + 1 < T_STEPS)       // stage xp[t+1] during GEMMs 1..8
          tdm_load_xp_tile(sxp_lds, xp + (size_t)(t + 1) * 64 * 512);
#if !defined(HAVE_TDM)
        // (fallback handled at end of timestep)
#endif
      }
      // prefetch first B fragment of the next GEMM
      {
        int nwm = (k == 8) ? 1 : (k + 2);
        __builtin_prefetch(packed + ((size_t)((nwm * 256 + p) * 32 + lane)) * 16, 0, 0);
      }
#pragma unroll
      for (int mt = 0; mt < 4; ++mt) {
#pragma unroll
        for (int r = 0; r < 8; ++r) {
          int row = mt * 16 + r + 8 * laneHi;
          int col = p * 16 + laneLo;
          float sn = snew[mt][r];
          if (k == 8) {
            sA[REG_SZ + row * SROW + col] = f2bf(sn);          // next h_prev
            out[((size_t)t * 64 + row) * 256 + col] = sn;
            if (t == T_STEPS - 1)
              out[((size_t)T_STEPS * 64 + row) * 256 + col] = sn;
          } else if (WREG[k] >= 0) {
            sA[WREG[k] * REG_SZ + row * SROW + col] = f2bf(sn);
          }
        }
      }
      __syncthreads();                          // writes visible before next GEMM
    }

    // make sure xp[t+1] stage is complete before next step's k=0 reads
#if defined(HAVE_TDM)
    if (wave == 0) tdm_wait0();
#else
    if (t + 1 < T_STEPS) {
      const unsigned short* src = xp + (size_t)(t + 1) * 64 * 512;
      for (int i = tid; i < 64 * 512; i += 512) sXP[i] = src[i];
    }
#endif
    __syncthreads();
  }
}

// ---------------------------------------------------------------------------
extern "C" void kernel_launch(void* const* d_in, const int* in_sizes, int n_in,
                              void* d_out, int out_size, void* d_ws, size_t ws_size,
                              hipStream_t stream) {
  const float* inputs = (const float*)d_in[0];   // [1024,64,256]
  const float* hidden = (const float*)d_in[1];   // [1,64,256]
  const float* W0     = (const float*)d_in[2];   // [512,512]
  const float* Ws     = (const float*)d_in[3];   // [8,256,512]
  float* out = (float*)d_out;

  unsigned short* packed = (unsigned short*)d_ws;           // 2.62 MB bf16 packed weights
  unsigned short* xp     = packed + PACKED_ELEMS;           // [1024,64,512] bf16 xproj

  pack_weights<<<5120, 256, 0, stream>>>(W0, Ws, packed);
  xproj_kernel<<<T_STEPS, 256, 0, stream>>>(inputs, packed, xp);
  recurrence<<<1, 512, 0, stream>>>(hidden, packed, xp, out);
}
